// SMPLStyledConv2d_446676599374
// MI455X (gfx1250) — matrix operand order repaired and checked by reading, not compiled, not run on hardware
//
#include <hip/hip_runtime.h>
#include <hip/hip_bf16.h>

// ---------------------------------------------------------------------------
// SMPL-styled modulated 3x3 conv, implicit GEMM on CDNA5 WMMA (bf16 -> f32)
//   B=8, Cin=256, Cout=256, H=W=128, K=3, SD=512, PD=170
// ---------------------------------------------------------------------------

typedef __attribute__((ext_vector_type(16))) __bf16 v16bf;
typedef __attribute__((ext_vector_type(8)))  __bf16 v8bf;
typedef __attribute__((ext_vector_type(8)))  float  v8f;

#define B_   8
#define CIN  256
#define COUT 256
#define H_   128
#define W_   128
#define PD_  170
#define SD_  512
#define KCH  32            // cin chunk per WMMA k-step
#define CONV_THREADS 512   // 16 waves (wave32)
#define BPAD 40            // padded cin stride in B tile (16B-aligned, conflict-friendly)

__device__ __forceinline__ unsigned short f2bf(float f) {
    unsigned u = __builtin_bit_cast(unsigned, f);
    unsigned r = u + 0x7FFFu + ((u >> 16) & 1u);   // round-to-nearest-even
    return (unsigned short)(r >> 16);
}

__device__ __forceinline__ v16bf cat16(v8bf lo, v8bf hi) {
    return __builtin_shufflevector(lo, hi, 0, 1, 2, 3, 4, 5, 6, 7,
                                           8, 9, 10, 11, 12, 13, 14, 15);
}

// ---------------------------------------------------------------------------
// Kernel 1: gamma/beta (SPADE) and s (style modulation), [B, Cin] each
// ---------------------------------------------------------------------------
__global__ void mod_params_kernel(const float* __restrict__ smpl,
                                  const float* __restrict__ style,
                                  const float* __restrict__ Wg,
                                  const float* __restrict__ bg,
                                  const float* __restrict__ Wb,
                                  const float* __restrict__ bb,
                                  const float* __restrict__ Wmod,
                                  const float* __restrict__ bmod,
                                  float* __restrict__ gamma,
                                  float* __restrict__ beta,
                                  float* __restrict__ s) {
    const int b = blockIdx.x;      // 0..7
    const int c = threadIdx.x;     // 0..255
    const float* sp = smpl + b * PD_;
    float gsum = 0.f, bsum = 0.f;
    for (int d = 0; d < PD_; ++d) {
        float sv = sp[d];
        gsum += sv * Wg[c * PD_ + d];
        bsum += sv * Wb[c * PD_ + d];
    }
    const float* st = style + b * SD_;
    const float mod_scale = 0.04419417382415922f;   // 1/sqrt(512)
    float ssum = 0.f;
    for (int d = 0; d < SD_; ++d) ssum += st[d] * Wmod[c * SD_ + d];
    gamma[b * CIN + c] = gsum + bg[c];
    beta [b * CIN + c] = bsum + bb[c];
    s    [b * CIN + c] = ssum * mod_scale + bmod[c];
}

// ---------------------------------------------------------------------------
// Kernel 2: modulate + demodulate weights -> bf16, layout w2[b][kk][cout][cin]
// ---------------------------------------------------------------------------
__global__ void mod_weight_kernel(const float* __restrict__ weight,
                                  const float* __restrict__ s,
                                  unsigned short* __restrict__ w2) {
    const int cout = blockIdx.x;   // 0..255
    const int b    = blockIdx.y;   // 0..7
    const int t    = threadIdx.x;  // 0..255
    __shared__ float s_sh[CIN];
    __shared__ float red[256];
    s_sh[t] = s[b * CIN + t];
    __syncthreads();

    const float w_scale = 1.0f / 48.0f;   // 1/sqrt(Cin*K*K)
    float vals[9];
    int   idxs[9];
    float sq = 0.f;
#pragma unroll
    for (int j = 0; j < 9; ++j) {
        int idx = t + j * 256;            // 0..2303 == cin*9+kk, coalesced
        float wv = w_scale * weight[cout * (CIN * 9) + idx] * s_sh[idx / 9];
        vals[j] = wv;
        idxs[j] = idx;
        sq += wv * wv;
    }
    red[t] = sq;
    __syncthreads();
    for (int off = 128; off > 0; off >>= 1) {
        if (t < off) red[t] += red[t + off];
        __syncthreads();
    }
    const float demod = rsqrtf(red[0] + 1e-8f);
#pragma unroll
    for (int j = 0; j < 9; ++j) {
        int cin = idxs[j] / 9, kk = idxs[j] % 9;
        w2[(((size_t)b * 9 + kk) * COUT + cout) * CIN + cin] = f2bf(vals[j] * demod);
    }
}

// ---------------------------------------------------------------------------
// Kernel 3: implicit-GEMM 3x3 conv.  One block per (y, b); block tile
// [Cout=256 x W=128].  Per cin-chunk: stage 3 input rows + all nine
// (ky,kx) weight slices in LDS (~181 KB of the WGP's 320 KB), then run
// 72 barrier-free WMMAs per wave.
// ---------------------------------------------------------------------------
__launch_bounds__(CONV_THREADS)
__global__ void conv_wmma_kernel(const float* __restrict__ x,
                                 const float* __restrict__ gamma,
                                 const float* __restrict__ beta,
                                 const unsigned short* __restrict__ w2,
                                 float* __restrict__ out) {
    const int y    = blockIdx.x;          // output row
    const int b    = blockIdx.y;          // batch
    const int tid  = threadIdx.x;
    const int lane = tid & 31;
    const int wave = tid >> 5;            // 0..15
    const int waveM0 = (wave & 3) * 64;   // cout tile origin (4 waves in M)
    const int waveN0 = (wave >> 2) * 32;  // pixel tile origin (4 waves in N)

    // A tiles: 9 x [cout=256][cin=32] bf16 (144 KB)
    __shared__ __align__(16) unsigned short Alds[9 * COUT * KCH];
    // B tile: [dy=3][col=130][cin=32 pad 40] bf16 (~30.5 KB); col 0 == x=-1
    __shared__ __align__(16) unsigned short Blds[3 * 130 * BPAD];
    __shared__ float gsh[CIN];
    __shared__ float bsh[CIN];

    // ---- block prologue: cache gamma/beta; zero the constant pad columns
    if (tid < 256)       gsh[tid]       = gamma[b * CIN + tid];
    else                 bsh[tid - 256] = beta [b * CIN + tid - 256];
    if (tid < 256 + CIN) { /* nothing */ }
    if (tid < 256)       bsh[tid]       = beta [b * CIN + tid];   // full fill
    for (int i = tid; i < 3 * KCH * 2; i += CONV_THREADS) {       // 192 zeros
        int dy  = i / (KCH * 2);
        int rem = i % (KCH * 2);
        int col = (rem & 1) ? 129 : 0;
        int c   = rem >> 1;
        Blds[(dy * 130 + col) * BPAD + c] = 0;
    }

    v8f acc[4][2] = {};

    const int nlo   = lane & 15;          // M (A) or N (B/C) index within 16
    const int halfA = (lane >> 4) * 8;    // A: K = halfA+e (e<8), 16+halfA+(e-8)
    const int halfB = (lane >> 4) * 8;    // B: K = halfB+e (e<8), 16+halfB+(e-8)

    for (int c0 = 0; c0 < CIN; c0 += KCH) {
        __syncthreads();   // previous chunk's fragment reads complete

        // ---- stage B: interior cols 1..128 (input x 0..127), float4 loads
        for (int i = tid; i < 3 * KCH * 32; i += CONV_THREADS) {
            int q  = i & 31;               // x quad: cols q*4 .. q*4+3
            int pc = i >> 5;               // 0..95
            int dy = pc % 3;
            int c  = pc / 3;               // 0..31
            int iy = y + dy - 1;
            int base = (dy * 130 + 1 + q * 4) * BPAD + c;
            if ((unsigned)iy < (unsigned)H_) {
                const float4 v = *(const float4*)
                    &x[(((size_t)b * CIN + c0 + c) * H_ + iy) * W_ + q * 4];
                float g = gsh[c0 + c], be = bsh[c0 + c];
                Blds[base]            = f2bf(v.x * g + be);
                Blds[base + BPAD]     = f2bf(v.y * g + be);
                Blds[base + 2 * BPAD] = f2bf(v.z * g + be);
                Blds[base + 3 * BPAD] = f2bf(v.w * g + be);
            } else {                        // zero-pad rows (post-modulation pad)
                Blds[base]            = 0;
                Blds[base + BPAD]     = 0;
                Blds[base + 2 * BPAD] = 0;
                Blds[base + 3 * BPAD] = 0;
            }
        }
        // ---- stage A: all nine (ky,kx) slices [256 x 32] for this chunk
        {
            const unsigned short* src =
                w2 + ((size_t)b * 9 * COUT) * CIN + c0;
            for (int i = tid; i < 9 * 1024; i += CONV_THREADS) {  // 16B units
                int kk  = i >> 10;
                int r   = i & 1023;
                int row = r >> 2;           // cout
                int seg = r & 3;            // 8 bf16 each
                *(uint4*)&Alds[kk * (COUT * KCH) + row * KCH + seg * 8] =
                    *(const uint4*)&src[((size_t)kk * COUT + row) * CIN + seg * 8];
            }
        }
        __syncthreads();

        // ---- 72 WMMAs per wave, no barriers
#pragma unroll
        for (int kk = 0; kk < 9; ++kk) {
            const int ky = kk / 3, kx = kk % 3;
            const unsigned short* Ak = &Alds[kk * (COUT * KCH)];

            v16bf af[4];
#pragma unroll
            for (int fm = 0; fm < 4; ++fm) {
                int row = waveM0 + fm * 16 + nlo;
                af[fm] = cat16(*(const v8bf*)&Ak[row * KCH + halfA],
                               *(const v8bf*)&Ak[row * KCH + 16 + halfA]);
            }
            v16bf bfr[2];
#pragma unroll
            for (int fn = 0; fn < 2; ++fn) {
                int base = ((ky * 130 + waveN0 + fn * 16 + nlo + kx) * BPAD) + halfB;
                bfr[fn] = cat16(*(const v8bf*)&Blds[base],
                                *(const v8bf*)&Blds[base + 16]);
            }
#pragma unroll
            for (int fm = 0; fm < 4; ++fm)
#pragma unroll
                for (int fn = 0; fn < 2; ++fn)
                    acc[fm][fn] = __builtin_amdgcn_wmma_f32_16x16x32_bf16(
                        false, af[fm], false, bfr[fn],
                        (short)0, acc[fm][fn], false, false);
        }
    }

    // ---- epilogue: C/D layout -> lanes 0-15: M=r, lanes 16-31: M=8+r; N=lane%16
    const int mhi = (lane >> 4) * 8;
#pragma unroll
    for (int fm = 0; fm < 4; ++fm) {
#pragma unroll
        for (int fn = 0; fn < 2; ++fn) {
            int xcol = waveN0 + fn * 16 + nlo;
#pragma unroll
            for (int r = 0; r < 8; ++r) {
                int co = waveM0 + fm * 16 + mhi + r;
                out[(((size_t)b * COUT + co) * H_ + y) * W_ + xcol] = acc[fm][fn][r];
            }
        }
    }
}

// ---------------------------------------------------------------------------
extern "C" void kernel_launch(void* const* d_in, const int* in_sizes, int n_in,
                              void* d_out, int out_size, void* d_ws, size_t ws_size,
                              hipStream_t stream) {
    (void)in_sizes; (void)n_in; (void)out_size; (void)ws_size;
    const float* input  = (const float*)d_in[0];
    const float* smpl   = (const float*)d_in[1];
    const float* style  = (const float*)d_in[2];
    const float* weight = (const float*)d_in[3];
    const float* Wg     = (const float*)d_in[4];
    const float* bg     = (const float*)d_in[5];
    const float* Wb     = (const float*)d_in[6];
    const float* bb     = (const float*)d_in[7];
    const float* Wmod   = (const float*)d_in[8];
    const float* bmod   = (const float*)d_in[9];
    float* out = (float*)d_out;

    // workspace layout: gamma/beta/s (fp32) then bf16 weights at +32KB
    float* gamma = (float*)d_ws;
    float* beta  = gamma + B_ * CIN;
    float* s     = beta  + B_ * CIN;
    unsigned short* w2 = (unsigned short*)((char*)d_ws + 32 * 1024);  // ~9.44 MB

    mod_params_kernel<<<dim3(B_), dim3(CIN), 0, stream>>>(
        smpl, style, Wg, bg, Wb, bb, Wmod, bmod, gamma, beta, s);

    mod_weight_kernel<<<dim3(COUT, B_), dim3(256), 0, stream>>>(weight, s, w2);

    conv_wmma_kernel<<<dim3(H_, B_), dim3(CONV_THREADS), 0, stream>>>(
        input, gamma, beta, w2, out);
}
